// SelfAttention_39238821216745
// MI455X (gfx1250) — matrix operand order repaired and checked by reading, not compiled, not run on hardware
//
#include <hip/hip_runtime.h>

typedef __attribute__((ext_vector_type(16))) _Float16 v16h;
typedef __attribute__((ext_vector_type(8)))  float    v8f;

#define B_  2
#define L_  2048
#define E_  1024
#define H_  16
#define D_  64
#define TOK 4096          // B_*L_
#define LDA 40            // padded lda (halfs) for 32-wide LDS tiles; 80B rows keep 16B align
#define LDK 72            // padded lda (halfs) for 64-wide K tile; 144B rows keep 16B align

// ---------------------------------------------------------------- helpers ---

__device__ __forceinline__ v8f wmma16(v16h a, v16h b, v8f c) {
  // D = A(16x32 f16) * B(32x16 f16) + C(16x16 f32)
  return __builtin_amdgcn_wmma_f32_16x16x32_f16(false, a, false, b, (short)0, c,
                                                false, false);
}

// A-matrix fragment (16x32 f16): lane holds row m=lane&15.
// halves 0-7  : K = hi*8   .. hi*8+7
// halves 8-15 : K = 16+hi*8 .. 16+hi*8+7       (hi = lane>>4)
__device__ __forceinline__ v16h ld_frag_a(const _Float16* rowp, int hi) {
  v16h a;
  ((uint4*)&a)[0] = *(const uint4*)(rowp + hi * 8);
  ((uint4*)&a)[1] = *(const uint4*)(rowp + 16 + hi * 8);
  return a;
}

// B-matrix fragment (32x16 f16): lane holds column n=lane&15.
// halves 0-7  : K = hi*8   .. hi*8+7
// halves 8-15 : K = 16+hi*8 .. 16+hi*8+7       (hi = lane>>4)
__device__ __forceinline__ v16h ld_frag_b(const _Float16* colp, int hi) {
  v16h b;
  ((uint4*)&b)[0] = *(const uint4*)(colp + hi * 8);
  ((uint4*)&b)[1] = *(const uint4*)(colp + 16 + hi * 8);
  return b;
}

// LDS byte address of a generic __shared__ pointer: the flat->LDS mapping is
// truncation to the low 32 bits (ISA 10.2).
__device__ __forceinline__ unsigned lds_addr(const void* p) {
  return (unsigned)(size_t)p;
}

// Async global -> LDS copy, 16 bytes per lane per instruction (ASYNCcnt).
// GVS addressing: mem = SADDR(64) + VADDR(32) + INST_OFFSET; the instruction
// offset is applied to BOTH the LDS and global addresses, so a second issue
// with offset:16 copies the adjacent 16B chunk with the same registers.
__device__ __forceinline__ void async_copy_32B(unsigned ldsOff, unsigned gOff,
                                               const void* gBase) {
  asm volatile(
      "global_load_async_to_lds_b128 %0, %1, %2\n\t"
      "global_load_async_to_lds_b128 %0, %1, %2 offset:16"
      :
      : "v"(ldsOff), "v"(gOff), "s"(gBase)
      : "memory");
}

__device__ __forceinline__ void async_copy_16B(unsigned ldsOff, unsigned gOff,
                                               const void* gBase) {
  asm volatile("global_load_async_to_lds_b128 %0, %1, %2"
               :
               : "v"(ldsOff), "v"(gOff), "s"(gBase)
               : "memory");
}

// Partial wait: async loads complete in order, so waiting down to the size of
// the most recently issued batch guarantees all earlier batches are resident.
template <int N>
__device__ __forceinline__ void wait_async_le() {
  asm volatile("s_wait_asynccnt %0" ::"i"(N) : "memory");
}

__device__ __forceinline__ float fixnum(float v) {
  v = (v != v) ? 0.0f : v;
  v = fminf(v, 1e9f);
  v = fmaxf(v, -1e9f);
  return v;
}

// ------------------------------------------------------------- converters ---

__global__ __launch_bounds__(256) void cvt_fix_f16(const float* __restrict__ in,
                                                   _Float16* __restrict__ out,
                                                   int n) {
  int i = blockIdx.x * 256 + threadIdx.x;
  if (i < n) out[i] = (_Float16)fixnum(in[i]);
}

__global__ __launch_bounds__(256) void cvt_f16(const float* __restrict__ in,
                                               _Float16* __restrict__ out,
                                               int n) {
  int i = blockIdx.x * 256 + threadIdx.x;
  if (i < n) out[i] = (_Float16)in[i];
}

// ------------------------------------------------------------------ GEMM ----
// C[t, j] = sum_e A[t, e] * W[j, e]   (torch Linear: x @ W.T)
// mode 0/1/2 : +bias, (mode 0: *0.125 Q-scale), store f16 into (B,H,L,D)
// mode 3     : +bias + residual x (nan-fixed), store f32 row-major [t][j]
// Double-buffered async LDS staging: slab k+1 flies while slab k is in WMMA.

__global__ __launch_bounds__(256) void gemm128(
    const _Float16* __restrict__ A, const _Float16* __restrict__ W,
    const float* __restrict__ bias, _Float16* __restrict__ outH,
    float* __restrict__ outF, const float* __restrict__ xres, int mode) {
  __shared__ _Float16 As[2][128 * LDA];
  __shared__ _Float16 Bs[2][128 * LDA];

  const int tid   = threadIdx.x;
  const int nBase = blockIdx.x * 128;
  const int mBase = blockIdx.y * 128;
  const int wid   = tid >> 5;
  const int lane  = tid & 31;
  const int ln    = lane & 15;
  const int hi    = lane >> 4;
  const int wm    = (wid >> 2) * 64;   // wave M offset within 128
  const int wn    = (wid & 3) * 32;    // wave N offset within 128

  const v8f vzero = {0.f, 0.f, 0.f, 0.f, 0.f, 0.f, 0.f, 0.f};
  v8f acc[4][2];
#pragma unroll
  for (int i = 0; i < 4; ++i)
#pragma unroll
    for (int j = 0; j < 2; ++j) acc[i][j] = vzero;

  const int srow = tid >> 1;            // 0..127
  const int sseg = (tid & 1) * 16;      // 0 or 16 halfs
  const unsigned ldsA0 = lds_addr(&As[0][srow * LDA + sseg]);
  const unsigned ldsA1 = lds_addr(&As[1][srow * LDA + sseg]);
  const unsigned ldsB0 = lds_addr(&Bs[0][srow * LDA + sseg]);
  const unsigned ldsB1 = lds_addr(&Bs[1][srow * LDA + sseg]);
  const unsigned gRowA = (unsigned)(((mBase + srow) * E_ + sseg) * 2);
  const unsigned gRowB = (unsigned)(((nBase + srow) * E_ + sseg) * 2);

  // prologue: batch 0 -> buffer 0
  async_copy_32B(ldsA0, gRowA, A);
  async_copy_32B(ldsB0, gRowB, W);

  for (int kb = 0; kb < E_ / 32; ++kb) {
    const int cur = kb & 1;
    if (kb + 1 < E_ / 32) {
      // kick batch kb+1 into the other buffer, then drain batch kb only
      const unsigned kOff = (unsigned)((kb + 1) * 64);  // bytes
      async_copy_32B(cur ? ldsA0 : ldsA1, gRowA + kOff, A);
      async_copy_32B(cur ? ldsB0 : ldsB1, gRowB + kOff, W);
      wait_async_le<4>();
    } else {
      wait_async_le<0>();
    }
    if (kb + 2 < E_ / 32) {  // pull slab kb+2 toward the WGP
      __builtin_prefetch(A + (size_t)(mBase + srow) * E_ + (kb + 2) * 32 + sseg, 0, 3);
      __builtin_prefetch(W + (size_t)(nBase + srow) * E_ + (kb + 2) * 32 + sseg, 0, 3);
    }
    __syncthreads();

    const _Float16* Asb = As[cur];
    const _Float16* Bsb = Bs[cur];
    v16h bf0 = ld_frag_b(Bsb + (wn + ln) * LDA, hi);
    v16h bf1 = ld_frag_b(Bsb + (wn + 16 + ln) * LDA, hi);
#pragma unroll
    for (int tm = 0; tm < 4; ++tm) {
      v16h af = ld_frag_a(Asb + (wm + tm * 16 + ln) * LDA, hi);
      acc[tm][0] = wmma16(af, bf0, acc[tm][0]);
      acc[tm][1] = wmma16(af, bf1, acc[tm][1]);
    }
    __syncthreads();
  }

  const float qscale = (mode == 0) ? 0.125f : 1.0f;
#pragma unroll
  for (int tm = 0; tm < 4; ++tm)
#pragma unroll
    for (int tn = 0; tn < 2; ++tn)
#pragma unroll
      for (int r = 0; r < 8; ++r) {
        const int t = mBase + wm + tm * 16 + hi * 8 + r;   // token
        const int j = nBase + wn + tn * 16 + ln;           // out feature
        float val = acc[tm][tn][r] + bias[j];
        if (mode < 3) {
          val *= qscale;
          const int h = j >> 6, d = j & 63;
          const int b = t >> 11, l = t & (L_ - 1);
          outH[(((size_t)b * H_ + h) * L_ + l) * D_ + d] = (_Float16)val;
        } else {
          const size_t idx = (size_t)t * E_ + j;
          outF[idx] = val + fixnum(xres[idx]);
        }
      }
}

// ------------------------------------------------------- flash attention ----
// One workgroup = 128 query rows of one (b,h); one wave = 16 query rows.
// Q pre-scaled by 1/sqrt(D). Causal mask, online softmax in registers.
// K tiles double-buffered via async LDS copies; V pipelined through VGPRs
// one block ahead (transpose scatter), so global latency hides under WMMA.

__global__ __launch_bounds__(256) void attn128(const _Float16* __restrict__ qh,
                                               const _Float16* __restrict__ kh,
                                               const _Float16* __restrict__ vh,
                                               _Float16* __restrict__ oh) {
  __shared__ _Float16 Ks[2][32 * LDK];     // [key][d]
  __shared__ _Float16 Vt[2][64 * LDA];     // [d][key]  (transposed V)
  __shared__ _Float16 Ps[8][16 * LDA];     // per-wave P tile [m][key]

  const int tid   = threadIdx.x;
  const int bh    = blockIdx.y;            // b*H + h
  const int qBase = blockIdx.x * 128;
  const int wid   = tid >> 5;
  const int lane  = tid & 31;
  const int ln    = lane & 15;
  const int hi    = lane >> 4;
  const int tBase = qBase + wid * 16;

  const _Float16* kBase = kh + (size_t)bh * L_ * D_;   // uniform per block
  const _Float16* vBase = vh + (size_t)bh * L_ * D_;

  // Q fragments for this wave's 16 rows (row = tBase+ln, K dim = d)
  const _Float16* qp = qh + ((size_t)bh * L_ + tBase + ln) * D_;
  const v16h qa0 = ld_frag_a(qp, hi);        // d = 0..31
  const v16h qa1 = ld_frag_a(qp + 32, hi);   // d = 32..63

  const v8f vzero = {0.f, 0.f, 0.f, 0.f, 0.f, 0.f, 0.f, 0.f};
  v8f oacc[4];
#pragma unroll
  for (int tn = 0; tn < 4; ++tn) oacc[tn] = vzero;
  float mrow[8], srow[8];
#pragma unroll
  for (int r = 0; r < 8; ++r) { mrow[r] = -1e30f; srow[r] = 0.0f; }

  const int nSteps = (qBase >> 5) + 4;      // key blocks needed (causal)
  const int kr = tid >> 3;                  // 0..31 key row for staging
  const int dg = (tid & 7) * 8;             // d-group 0,8,...,56
  const unsigned ldsK0 = lds_addr(&Ks[0][kr * LDK + dg]);
  const unsigned ldsK1 = lds_addr(&Ks[1][kr * LDK + dg]);

  // prologue: batch 0 (K async into buf0; V block 0 into registers)
  async_copy_16B(ldsK0, (unsigned)((kr * D_ + dg) * 2), kBase);
  uint4 vv = *(const uint4*)(vBase + (size_t)kr * D_ + dg);

  for (int s = 0; s < nSteps; ++s) {
    const int cur = s & 1;
    const int sBase = s * 32;
    {
      // scatter V block s (loaded one iteration ago) into Vt[cur]
      const _Float16* v8p = (const _Float16*)&vv;
      _Float16* vtb = Vt[cur];
#pragma unroll
      for (int j = 0; j < 8; ++j) vtb[(dg + j) * LDA + kr] = v8p[j];
    }
    if (s + 1 < nSteps) {
      // kick batch s+1: K async into other buffer, V into registers
      const unsigned gOffK = (unsigned)(((sBase + 32 + kr) * D_ + dg) * 2);
      async_copy_16B(cur ? ldsK0 : ldsK1, gOffK, kBase);
      vv = *(const uint4*)(vBase + (size_t)(sBase + 32 + kr) * D_ + dg);
      wait_async_le<1>();
    } else {
      wait_async_le<0>();
    }
    __syncthreads();

    if (sBase <= tBase + 15) {
      const _Float16* Ksb = Ks[cur];
      const _Float16* Vtb = Vt[cur];
      // S = Q * K^T  (two 16x16 tiles covering 32 keys)
      v8f s0 = vzero, s1 = vzero;
      s0 = wmma16(qa0, ld_frag_b(Ksb + ln * LDK, hi), s0);
      s0 = wmma16(qa1, ld_frag_b(Ksb + ln * LDK + 32, hi), s0);
      s1 = wmma16(qa0, ld_frag_b(Ksb + (16 + ln) * LDK, hi), s1);
      s1 = wmma16(qa1, ld_frag_b(Ksb + (16 + ln) * LDK + 32, hi), s1);

#pragma unroll
      for (int r = 0; r < 8; ++r) {
        const int t = tBase + hi * 8 + r;
        float e0 = (sBase + ln <= t) ? s0[r] : -1e30f;
        float e1 = (sBase + 16 + ln <= t) ? s1[r] : -1e30f;
        float mx = fmaxf(e0, e1);
        mx = fmaxf(mx, __shfl_xor(mx, 1, 32));
        mx = fmaxf(mx, __shfl_xor(mx, 2, 32));
        mx = fmaxf(mx, __shfl_xor(mx, 4, 32));
        mx = fmaxf(mx, __shfl_xor(mx, 8, 32));
        const float mnew = fmaxf(mrow[r], mx);
        const float corr = __expf(mrow[r] - mnew);
        const float p0 = __expf(e0 - mnew);
        const float p1 = __expf(e1 - mnew);
        float rs = p0 + p1;
        rs += __shfl_xor(rs, 1, 32);
        rs += __shfl_xor(rs, 2, 32);
        rs += __shfl_xor(rs, 4, 32);
        rs += __shfl_xor(rs, 8, 32);
        srow[r] = srow[r] * corr + rs;
        mrow[r] = mnew;
#pragma unroll
        for (int tn = 0; tn < 4; ++tn) oacc[tn][r] *= corr;
        // P (C-layout) -> LDS, to be re-read in A-layout
        Ps[wid][(hi * 8 + r) * LDA + ln] = (_Float16)p0;
        Ps[wid][(hi * 8 + r) * LDA + 16 + ln] = (_Float16)p1;
      }

      // O += P * V   (four 16x16 output tiles over d)
      const v16h pa = ld_frag_a(&Ps[wid][ln * LDA], hi);
#pragma unroll
      for (int tn = 0; tn < 4; ++tn) {
        const v16h vf = ld_frag_b(Vtb + (tn * 16 + ln) * LDA, hi);
        oacc[tn] = wmma16(pa, vf, oacc[tn]);
      }
    }
    __syncthreads();
  }

  // Normalize and store token-major f16: oh[token][h*64+d]
  const int b = bh >> 4, h = bh & 15;
#pragma unroll
  for (int r = 0; r < 8; ++r) {
    const float inv = 1.0f / srow[r];
    const size_t token = (size_t)b * L_ + tBase + hi * 8 + r;
#pragma unroll
    for (int tn = 0; tn < 4; ++tn) {
      const int d = tn * 16 + ln;
      oh[token * E_ + h * D_ + d] = (_Float16)(oacc[tn][r] * inv);
    }
  }
}

// ------------------------------------------------------------- LayerNorm ----

__global__ __launch_bounds__(256) void layernorm_k(const float* __restrict__ rr,
                                                   const float* __restrict__ gamma,
                                                   const float* __restrict__ beta,
                                                   float* __restrict__ out) {
  __shared__ float s1[256], s2[256];
  const int row = blockIdx.x;
  const int tid = threadIdx.x;
  const float4 v = *(const float4*)(rr + (size_t)row * E_ + tid * 4);
  s1[tid] = v.x + v.y + v.z + v.w;
  s2[tid] = v.x * v.x + v.y * v.y + v.z * v.z + v.w * v.w;
  __syncthreads();
  for (int off = 128; off > 0; off >>= 1) {
    if (tid < off) {
      s1[tid] += s1[tid + off];
      s2[tid] += s2[tid + off];
    }
    __syncthreads();
  }
  const float mu = s1[0] * (1.0f / E_);
  const float var = s2[0] * (1.0f / E_) - mu * mu;
  const float rstd = rsqrtf(var + 1e-5f);
  const float4 g = *(const float4*)(gamma + tid * 4);
  const float4 bt = *(const float4*)(beta + tid * 4);
  float4 o;
  o.x = fixnum((v.x - mu) * rstd * g.x + bt.x);
  o.y = fixnum((v.y - mu) * rstd * g.y + bt.y);
  o.z = fixnum((v.z - mu) * rstd * g.z + bt.z);
  o.w = fixnum((v.w - mu) * rstd * g.w + bt.w);
  *(float4*)(out + (size_t)row * E_ + tid * 4) = o;
}

// ----------------------------------------------------------------- launch ---

extern "C" void kernel_launch(void* const* d_in, const int* in_sizes, int n_in,
                              void* d_out, int out_size, void* d_ws, size_t ws_size,
                              hipStream_t stream) {
  (void)in_sizes; (void)n_in; (void)out_size; (void)ws_size;
  const float* x     = (const float*)d_in[0];
  // d_in[1] = bool causal mask (implemented analytically)
  const float* Wq    = (const float*)d_in[2];
  const float* bq    = (const float*)d_in[3];
  const float* Wk    = (const float*)d_in[4];
  const float* bk    = (const float*)d_in[5];
  const float* Wv    = (const float*)d_in[6];
  const float* bv    = (const float*)d_in[7];
  const float* Wo    = (const float*)d_in[8];
  const float* bo    = (const float*)d_in[9];
  const float* gamma = (const float*)d_in[10];
  const float* beta  = (const float*)d_in[11];
  float* out = (float*)d_out;

  _Float16* xh  = (_Float16*)d_ws;
  _Float16* wqh = xh + (size_t)TOK * E_;
  _Float16* wkh = wqh + (size_t)E_ * E_;
  _Float16* wvh = wkh + (size_t)E_ * E_;
  _Float16* woh = wvh + (size_t)E_ * E_;
  _Float16* q   = woh + (size_t)E_ * E_;
  _Float16* k   = q + (size_t)TOK * E_;
  _Float16* v   = k + (size_t)TOK * E_;
  _Float16* o   = v + (size_t)TOK * E_;
  float* rbuf   = (float*)(o + (size_t)TOK * E_);

  const int nTok = TOK * E_;
  const int nW = E_ * E_;
  cvt_fix_f16<<<nTok / 256, 256, 0, stream>>>(x, xh, nTok);
  cvt_f16<<<nW / 256, 256, 0, stream>>>(Wq, wqh, nW);
  cvt_f16<<<nW / 256, 256, 0, stream>>>(Wk, wkh, nW);
  cvt_f16<<<nW / 256, 256, 0, stream>>>(Wv, wvh, nW);
  cvt_f16<<<nW / 256, 256, 0, stream>>>(Wo, woh, nW);

  dim3 gg(E_ / 128, TOK / 128);
  gemm128<<<gg, 256, 0, stream>>>(xh, wqh, bq, q, nullptr, nullptr, 0);
  gemm128<<<gg, 256, 0, stream>>>(xh, wkh, bk, k, nullptr, nullptr, 1);
  gemm128<<<gg, 256, 0, stream>>>(xh, wvh, bv, v, nullptr, nullptr, 2);

  attn128<<<dim3(L_ / 128, B_ * H_), 256, 0, stream>>>(q, k, v, o);

  gemm128<<<gg, 256, 0, stream>>>(o, woh, bo, nullptr, rbuf, x, 3);
  layernorm_k<<<TOK, 256, 0, stream>>>(rbuf, gamma, beta, out);
}
